// EMACodebook_38774964748792
// MI455X (gfx1250) — compile-verified, matrix operand order checked by reading, not run on hardware
//
#include <hip/hip_runtime.h>
#include <hip/hip_bf16.h>

typedef __attribute__((ext_vector_type(16))) _Float16 v16h;
typedef __attribute__((ext_vector_type(8)))  float    v8f;

#define NROWS 8192   // B*T
#define DDIM  512
#define VCB   8192

union AFrag { v16h v; uint4 q[2]; };

// ---- CDNA5 async global->LDS copy (ASYNCcnt-tracked), 16B per lane ----------
__device__ __forceinline__ void async_copy_b128(unsigned int lds_addr, const void* gaddr) {
    // VDST = VGPR holding LDS byte address, VADDR = 64-bit global address (GV mode)
    asm volatile("global_load_async_to_lds_b128 %0, %1, off"
                 :: "v"(lds_addr), "v"(gaddr) : "memory");
}
__device__ __forceinline__ void wait_async0() {
#if __has_builtin(__builtin_amdgcn_s_wait_asynccnt)
    __builtin_amdgcn_s_wait_asynccnt(0);
#else
    asm volatile("s_wait_asynccnt 0x0" ::: "memory");
#endif
}

// ---------------------------------------------------------------- zero scratch
__global__ void k_zero(float* p, int n) {
    int i = blockIdx.x * blockDim.x + threadIdx.x;
    if (i < n) p[i] = 0.0f;
}

// --------------------------------------------------- normalize rows -> f16 copies
// rows [0, NROWS) come from x, rows [NROWS, NROWS+VCB) from codebook.
__global__ __launch_bounds__(256) void k_normalize(const float* __restrict__ x,
                                                   const float* __restrict__ cb,
                                                   _Float16* __restrict__ xn,
                                                   _Float16* __restrict__ cbn) {
    int row  = blockIdx.x * 8 + (threadIdx.x >> 5);
    int lane = threadIdx.x & 31;
    const float* src;
    _Float16*    dst;
    if (row < NROWS) { src = x  + (size_t)row * DDIM;           dst = xn  + (size_t)row * DDIM; }
    else             { int r = row - NROWS;
                       src = cb + (size_t)r   * DDIM;           dst = cbn + (size_t)r   * DDIM; }

    const float4* s4 = (const float4*)src;
    float4 v[4];
    float ss = 0.0f;
#pragma unroll
    for (int j = 0; j < 4; ++j) {
        v[j] = s4[lane + j * 32];
        ss += v[j].x * v[j].x + v[j].y * v[j].y + v[j].z * v[j].z + v[j].w * v[j].w;
    }
#pragma unroll
    for (int m = 1; m < 32; m <<= 1) ss += __shfl_xor(ss, m, 32);
    float inv = 1.0f / fmaxf(sqrtf(ss), 1e-12f);
#pragma unroll
    for (int j = 0; j < 4; ++j) {
        int base = (lane + j * 32) * 4;
        dst[base + 0] = (_Float16)(v[j].x * inv);
        dst[base + 1] = (_Float16)(v[j].y * inv);
        dst[base + 2] = (_Float16)(v[j].z * inv);
        dst[base + 3] = (_Float16)(v[j].w * inv);
    }
}

// --------------------------------------------------- WMMA cosine-sim argmax
// Block: 256 threads = 8 waves; each wave owns a 16-row strip (128 rows/block).
// A (16x512 f16) lives in registers; codebook streamed through double-buffered
// LDS in 32-row chunks via async global->LDS copies overlapped with WMMA.
#define LDS_STRIDE 520                    // 512+8 f16 pad: 1040B pitch, 4-bank rotation
#define CHUNK_ELEMS (32 * LDS_STRIDE)     // one 32-row chunk (f16 elements)
__global__ __launch_bounds__(256) void k_argmax(const _Float16* __restrict__ xn,
                                                const _Float16* __restrict__ cbn,
                                                int* __restrict__ best_out) {
    __shared__ _Float16 sB[2 * CHUNK_ELEMS];   // double buffer: 2 x 33,280 B

    const int tid     = threadIdx.x;
    const int lane    = tid & 31;
    const int wave    = tid >> 5;
    const int l15     = lane & 15;
    const int halfsel = lane >> 4;                 // 0: lanes 0-15, 1: lanes 16-31
    const int mrow    = blockIdx.x * 128 + wave * 16 + l15;

    // LDS byte address of buffer base (generic LDS pointer: offset in low 32 bits)
    const unsigned int sB_addr = (unsigned int)(size_t)(&sB[0]);

    // ---- load A fragments (entire K=512) into registers: 16 k-blocks of K=32
    v16h a[16];
#pragma unroll
    for (int k = 0; k < 16; ++k) {
        const int kb = k * 32;
        AFrag u;
        u.q[0] = *(const uint4*)(xn + (size_t)mrow * DDIM + kb +      8 * halfsel);
        u.q[1] = *(const uint4*)(xn + (size_t)mrow * DDIM + kb + 16 + 8 * halfsel);
        a[k] = u.v;
    }

    float bv[8];
    int   bi[8];
#pragma unroll
    for (int r = 0; r < 8; ++r) { bv[r] = -3.0e38f; bi[r] = 0; }

    // ---- preload chunk 0 into buffer 0 (32 rows x 512 f16 = 2048 x 16B)
#pragma unroll
    for (int i = tid; i < 2048; i += 256) {
        int row = i >> 5, cu = i & 31;
        async_copy_b128(sB_addr + (unsigned)(row * LDS_STRIDE + cu * 8) * 2,
                        cbn + (size_t)row * DDIM + cu * 8);
    }

    int bufsel = 0;
    for (int c0 = 0; c0 < VCB; c0 += 32) {
        wait_async0();                       // own async writes for chunk c landed
        __syncthreads();                     // all waves done with other buffer + visibility

        if (c0 + 32 < VCB) {                 // overlap: issue chunk c+1 into other buffer
            const unsigned int dst = sB_addr + (unsigned)((bufsel ^ 1) * CHUNK_ELEMS) * 2;
#pragma unroll
            for (int i = tid; i < 2048; i += 256) {
                int row = i >> 5, cu = i & 31;
                async_copy_b128(dst + (unsigned)(row * LDS_STRIDE + cu * 8) * 2,
                                cbn + (size_t)(c0 + 32 + row) * DDIM + cu * 8);
            }
        }

        const _Float16* buf = sB + bufsel * CHUNK_ELEMS;
        v8f acc0 = {}, acc1 = {};
#pragma unroll
        for (int k = 0; k < 16; ++k) {
            const int kb = k * 32;
            AFrag b0, b1;
            const _Float16* r0 = buf + (l15)      * LDS_STRIDE;
            const _Float16* r1 = buf + (16 + l15) * LDS_STRIDE;
            b0.q[0] = *(const uint4*)(r0 + kb +      8 * halfsel);
            b0.q[1] = *(const uint4*)(r0 + kb + 16 + 8 * halfsel);
            b1.q[0] = *(const uint4*)(r1 + kb +      8 * halfsel);
            b1.q[1] = *(const uint4*)(r1 + kb + 16 + 8 * halfsel);
            acc0 = __builtin_amdgcn_wmma_f32_16x16x32_f16(false, a[k], false, b0.v,
                                                          (short)0, acc0, false, false);
            acc1 = __builtin_amdgcn_wmma_f32_16x16x32_f16(false, a[k], false, b1.v,
                                                          (short)0, acc1, false, false);
        }
        // running argmax: reg r holds row (r + 8*halfsel), column l15 of each tile
#pragma unroll
        for (int r = 0; r < 8; ++r) {
            float v0 = acc0[r]; int i0 = c0 + l15;
            if (v0 > bv[r] || (v0 == bv[r] && i0 < bi[r])) { bv[r] = v0; bi[r] = i0; }
            float v1 = acc1[r]; int i1 = c0 + 16 + l15;
            if (v1 > bv[r] || (v1 == bv[r] && i1 < bi[r])) { bv[r] = v1; bi[r] = i1; }
        }
        bufsel ^= 1;
    }

    // reduce across the 16 lanes that share each row (width-16 xor shuffle)
#pragma unroll
    for (int r = 0; r < 8; ++r) {
#pragma unroll
        for (int m = 1; m <= 8; m <<= 1) {
            float ov = __shfl_xor(bv[r], m, 16);
            int   oi = __shfl_xor(bi[r], m, 16);
            if (ov > bv[r] || (ov == bv[r] && oi < bi[r])) { bv[r] = ov; bi[r] = oi; }
        }
        if (l15 == 0)
            best_out[blockIdx.x * 128 + wave * 16 + r + 8 * halfsel] = bi[r];
    }
}

// --------------------------------------------------- gather z_q, loss, counts
__global__ __launch_bounds__(256) void k_gather(const float* __restrict__ x,
                                                const float* __restrict__ cb,
                                                const int* __restrict__ idx,
                                                float* __restrict__ out_zq,
                                                float* __restrict__ out_idx,
                                                float* __restrict__ loss,
                                                float* __restrict__ counts) {
    __shared__ float sred[256];
    const int row = blockIdx.x;
    const int tid = threadIdx.x;
    const int v   = idx[row];
    float s = 0.0f;
#pragma unroll
    for (int j = tid; j < DDIM; j += 256) {
        float q  = cb[(size_t)v * DDIM + j];
        float xv = x[(size_t)row * DDIM + j];
        out_zq[(size_t)row * DDIM + j] = q;   // x + sg(q - x) == q numerically
        float d = xv - q;
        s += d * d;
    }
    sred[tid] = s;
    __syncthreads();
    for (int step = 128; step > 0; step >>= 1) {
        if (tid < step) sred[tid] += sred[tid + step];
        __syncthreads();
    }
    if (tid == 0) {
        atomicAdd(loss, sred[0]);
        atomicAdd(&counts[v], 1.0f);
        out_idx[row] = (float)v;
    }
}

// --------------------------------------------------- scalars: loss/ppl/util
__global__ __launch_bounds__(256) void k_finalize(const float* __restrict__ counts,
                                                  const float* __restrict__ loss,
                                                  float* __restrict__ out3) {
    __shared__ float se[256];
    __shared__ float su[256];
    const int tid = threadIdx.x;
    float e = 0.0f, u = 0.0f;
    for (int i = tid; i < VCB; i += 256) {
        float c = counts[i];
        float p = c * (1.0f / (float)NROWS);
        e += p * logf(p + 1e-10f);
        u += (c > 0.0f) ? 1.0f : 0.0f;
    }
    se[tid] = e; su[tid] = u;
    __syncthreads();
    for (int step = 128; step > 0; step >>= 1) {
        if (tid < step) { se[tid] += se[tid + step]; su[tid] += su[tid + step]; }
        __syncthreads();
    }
    if (tid == 0) {
        out3[0] = loss[0] * (1.0f / ((float)NROWS * (float)DDIM)); // commit_loss
        out3[1] = expf(-se[0]);                                    // perplexity
        out3[2] = su[0] * (1.0f / (float)VCB);                     // utilization
    }
}

// ---------------------------------------------------------------- launcher
extern "C" void kernel_launch(void* const* d_in, const int* in_sizes, int n_in,
                              void* d_out, int out_size, void* d_ws, size_t ws_size,
                              hipStream_t stream) {
    const float* x  = (const float*)d_in[0];   // [8,1024,512]
    const float* cb = (const float*)d_in[1];   // [8192,512]

    // workspace layout
    char* ws = (char*)d_ws;
    _Float16* xn     = (_Float16*)(ws);                               // 8 MB
    _Float16* cbn    = (_Float16*)(ws + (size_t)NROWS * DDIM * 2);    // 8 MB
    int*      idx    = (int*)     (ws + (size_t)(NROWS + VCB) * DDIM * 2);
    float*    counts = (float*)   ((char*)idx + (size_t)NROWS * 4);
    float*    loss   = (float*)   ((char*)counts + (size_t)VCB * 4);

    float* out_zq  = (float*)d_out;                                   // [N*D]
    float* out_idx = out_zq + (size_t)NROWS * DDIM;                   // [N]
    float* out3    = out_idx + NROWS;                                 // 3 scalars

    // 1. zero counts + loss (V + 1 floats)
    k_zero<<<(VCB + 1 + 255) / 256, 256, 0, stream>>>(counts, VCB + 1);
    // 2. normalize (N + V rows, 8 rows/block)
    k_normalize<<<(NROWS + VCB) / 8, 256, 0, stream>>>(x, cb, xn, cbn);
    // 3. WMMA similarity + argmax (128 rows/block)
    k_argmax<<<NROWS / 128, 256, 0, stream>>>(xn, cbn, idx);
    // 4. gather + commit loss + bincount
    k_gather<<<NROWS, 256, 0, stream>>>(x, cb, idx, out_zq, out_idx, loss, counts);
    // 5. scalars
    k_finalize<<<1, 256, 0, stream>>>(counts, loss, out3);
}
